// LcrRot_12558484373639
// MI455X (gfx1250) — matrix operand order repaired and checked by reading, not compiled, not run on hardware
//
#include <hip/hip_runtime.h>
#include <hip/hip_bf16.h>
#include <math.h>

typedef __attribute__((ext_vector_type(16))) __bf16 v16bf;
typedef __attribute__((ext_vector_type(8)))  __bf16 v8bf;
typedef __attribute__((ext_vector_type(8)))  float  v8f;

#define BATCH 64
#define EDIM 300
#define HDIM 300
#define NG   1200          // 4*H gate width
#define KP   320           // K padded to 10 chunks of 32 (bf16 WMMA)
#define LDH  328           // LDS h row stride (bf16 elems), bank spread
#define LT_  16
#define LC_  96
#define NROWS_TOT 26624    // 2*64*16 + 4*64*96

struct WArgs { const float* w[6]; const float* b[6]; };
struct SArgs { const int* ids[3]; const int* lens[3]; };

__device__ __forceinline__ int chain_T(int c)      { return (c < 2) ? LT_ : LC_; }
__device__ __forceinline__ int chain_rows(int c)   { return (c < 2) ? (BATCH * LT_) : (BATCH * LC_); }
__device__ __forceinline__ int chain_rowoff(int c) {
  return (c < 2) ? c * BATCH * LT_ : 2 * BATCH * LT_ + (c - 2) * BATCH * LC_;
}

__device__ __forceinline__ float sigm(float x) { return 1.0f / (1.0f + expf(-x)); }
__device__ __forceinline__ int imin(int a, int b) { return a < b ? a : b; }

// A-fragment load: elems 0-7 at +off, elems 8-15 at +off+16 (ISA 16-bit A 16x32 layout)
__device__ __forceinline__ v16bf load_a_frag(const __bf16* rowk, int off) {
  v8bf lo = *(const v8bf*)(rowk + off);
  v8bf hi = *(const v8bf*)(rowk + off + 16);
  return __builtin_shufflevector(lo, hi, 0,1,2,3,4,5,6,7,8,9,10,11,12,13,14,15);
}

// ---------------- K1: transpose+convert weights to bf16 [N][KP], copy bias ----
__global__ void prep_weights_kernel(WArgs p, __bf16* __restrict__ WxT,
                                    __bf16* __restrict__ WhT,
                                    float* __restrict__ biasB) {
  int idx = blockIdx.x * blockDim.x + threadIdx.x;
  const int per = NG * KP;
  if (idx >= 6 * per) return;
  int c = idx / per, r = idx % per;
  int n = r / KP, k = r % KP;
  const float* w = p.w[c];               // [600][1200], rows 0..299 = x, 300..599 = h
  __bf16 xv = (__bf16)0.0f, hv = (__bf16)0.0f;
  if (k < EDIM) {
    xv = (__bf16)w[(size_t)k * NG + n];
    hv = (__bf16)w[(size_t)(EDIM + k) * NG + n];
  }
  WxT[idx] = xv;
  WhT[idx] = hv;
  if (k == 0) biasB[c * NG + n] = p.b[c][n];
}

// ---------------- K2: gather embeddings into bf16 A matrices (fw + reversed bw)
__global__ void prep_embed_kernel(SArgs s, const float* __restrict__ emb,
                                  __bf16* __restrict__ Ab) {
  int idx = blockIdx.x * blockDim.x + threadIdx.x;
  if (idx >= NROWS_TOT * KP) return;
  int row = idx / KP, k = idx % KP;
  int c = (row < 1024) ? 0 : (row < 2048) ? 1 : (row < 8192) ? 2 :
          (row < 14336) ? 3 : (row < 20480) ? 4 : 5;
  int lr = row - chain_rowoff(c);
  int T = chain_T(c);
  int b = lr / T, t = lr % T;
  int seg = (c < 2) ? 0 : (c < 4) ? 1 : 2;
  int len = s.lens[seg][b];
  int tt = (c & 1) ? ((t < len) ? (len - 1 - t) : t) : t;   // odd chains = backward
  __bf16 v = (__bf16)0.0f;
  if (k < EDIM) {
    int tok = s.ids[seg][b * T + tt];
    v = (__bf16)emb[(size_t)tok * EDIM + k];
  }
  Ab[idx] = v;
}

// ---------------- K3: parallel x@Wx + bias -> preact[rows][1200] (WMMA bf16) --
__global__ __launch_bounds__(256) void xw_gemm_kernel(
    const __bf16* __restrict__ Ab, const __bf16* __restrict__ WxT,
    const float* __restrict__ biasB, float* __restrict__ preact) {
  const int c = blockIdx.y;
  const int lane = threadIdx.x & 31;
  const int wv = threadIdx.x >> 5;
  const int mtiles = chain_rows(c) >> 4;            // 64 or 384
  const int nwt = mtiles * 19;                      // 19 groups of 4 n-tiles
  int wt = blockIdx.x * 8 + wv;
  if (wt >= nwt) return;                            // wave-uniform exit
  const int mt = wt / 19, n4 = wt % 19;
  const int row0 = mt * 16, n0 = n4 * 64;
  const int li   = lane & 15;
  const int aoff = (lane >> 4) << 3;                // 0 / 8
  const int boff = (lane >> 4) << 3;                // 0 / 8
  const __bf16* Wc = WxT + (size_t)c * NG * KP;

  int ncol[4];
  const __bf16* bptr[4];                            // hoisted B row pointers
  v8f acc[4];
#pragma unroll
  for (int j = 0; j < 4; ++j) {
    ncol[j] = n0 + j * 16 + li;
    int nc = imin(ncol[j], NG - 1);                 // v_min, no exec divergence
    bptr[j] = Wc + nc * KP + boff;
    float bv = biasB[c * NG + nc];
#pragma unroll
    for (int r = 0; r < 8; ++r) acc[j][r] = bv;
  }
  const __bf16* arow = Ab + (size_t)(chain_rowoff(c) + row0 + li) * KP;
#pragma unroll
  for (int kk = 0; kk < KP; kk += 32) {
    v16bf a = load_a_frag(arow + kk, aoff);
#pragma unroll
    for (int j = 0; j < 4; ++j) {
      v16bf bf = load_a_frag(bptr[j] + kk, 0);
      acc[j] = __builtin_amdgcn_wmma_f32_16x16x32_bf16(
          false, a, false, bf, (short)0, acc[j], false, false);
    }
  }
  float* prow = preact + (size_t)(chain_rowoff(c) + row0) * NG;
#pragma unroll
  for (int j = 0; j < 4; ++j) {
    if (ncol[j] < NG) {
#pragma unroll
      for (int r = 0; r < 8; ++r)
        prow[(size_t)(aoff + r) * NG + ncol[j]] = acc[j][r];
    }
  }
}

// ---------------- K4: persistent recurrent LSTM, one workgroup per chain ------
__global__ __launch_bounds__(1024) void lstm_chain_kernel(
    SArgs s, const __bf16* __restrict__ WhT, const float* __restrict__ preact,
    float* __restrict__ pooled, float* __restrict__ h_l, float* __restrict__ h_r) {
  __shared__ __bf16 hS[BATCH * LDH];
  __shared__ int lenS[BATCH];
  const int c   = blockIdx.x;
  const int seg = (c < 2) ? 0 : (c < 4) ? 1 : 2;
  const int T   = chain_T(c);
  const int tid = threadIdx.x;
  const int lane = tid & 31, wv = tid >> 5;
  for (int i = tid; i < BATCH * LDH; i += blockDim.x) hS[i] = (__bf16)0.0f;
  if (tid < BATCH) lenS[tid] = s.lens[seg][tid];
  __syncthreads();

  const __bf16* Wc   = WhT + (size_t)c * NG * KP;
  const float* pre_c = preact + (size_t)chain_rowoff(c) * NG;
  float* outp = (c < 4) ? h_l : h_r;
  const int fw = !(c & 1);
  const int coloff = fw ? 0 : HDIM;

  const int li   = lane & 15;
  const int off8 = (lane >> 4) << 3;
  const int boff = (lane >> 4) << 3;

  // ---- per-wave persistent block descriptors (time-invariant, hoisted) ----
  int bm[3], bcol[3];
  const __bf16* barow[3];                 // A-row LDS pointer per block
  const __bf16* bwp[3][4];                // B row pointers, 4 gates per block
  const float*  bpre[3];                  // preact base (nc folded in)
#pragma unroll
  for (int i = 0; i < 3; ++i) {
    int sblk = wv + i * 32;               // 76 blocks = 4 m-tiles x 19 h-tiles
    int sb = imin(sblk, 75);              // harmless dummy for inactive slots
    int m = sb & 3, ht = sb >> 2;
    int col = ht * 16 + li;
    int nc = imin(col, HDIM - 1);
    bm[i] = m;
    bcol[i] = col;
    barow[i] = &hS[(m * 16 + li) * LDH];
#pragma unroll
    for (int g = 0; g < 4; ++g) bwp[i][g] = Wc + (g * HDIM + nc) * KP + boff;
    bpre[i] = pre_c + nc;
  }

  float cst[3][8], psum[3][8];
#pragma unroll
  for (int i = 0; i < 3; ++i)
#pragma unroll
    for (int r = 0; r < 8; ++r) { cst[i][r] = 0.0f; psum[i][r] = 0.0f; }

  for (int t = 0; t < T; ++t) {
    float h2buf[3][8];
#pragma unroll
    for (int i = 0; i < 3; ++i) {
      int sblk = wv + i * 32;
      if (sblk < 76) {                    // wave-uniform guard
        const int m = bm[i], col = bcol[i];
        v8f ai, aj, af, ao;
#pragma unroll
        for (int r = 0; r < 8; ++r) {     // init gate accs from x-preact
          int b = m * 16 + off8 + r;
          const float* pr = bpre[i] + (size_t)(b * T + t) * NG;
          ai[r] = pr[0 * HDIM];
          aj[r] = pr[1 * HDIM];
          af[r] = pr[2 * HDIM];
          ao[r] = pr[3 * HDIM];
        }
        const __bf16* arow = barow[i];
        const __bf16* p0 = bwp[i][0];
        const __bf16* p1 = bwp[i][1];
        const __bf16* p2 = bwp[i][2];
        const __bf16* p3 = bwp[i][3];
#pragma unroll
        for (int kk = 0; kk < KP; kk += 32) {
          v16bf a  = load_a_frag(arow + kk, off8);
          v16bf b0 = load_a_frag(p0 + kk, 0);
          v16bf b1 = load_a_frag(p1 + kk, 0);
          v16bf b2 = load_a_frag(p2 + kk, 0);
          v16bf b3 = load_a_frag(p3 + kk, 0);
          ai = __builtin_amdgcn_wmma_f32_16x16x32_bf16(false, a, false, b0, (short)0, ai, false, false);
          aj = __builtin_amdgcn_wmma_f32_16x16x32_bf16(false, a, false, b1, (short)0, aj, false, false);
          af = __builtin_amdgcn_wmma_f32_16x16x32_bf16(false, a, false, b2, (short)0, af, false, false);
          ao = __builtin_amdgcn_wmma_f32_16x16x32_bf16(false, a, false, b3, (short)0, ao, false, false);
          if (kk + 32 < KP)
            __builtin_prefetch(p0 + kk + 32, 0, 1);
        }
#pragma unroll
        for (int r = 0; r < 8; ++r) {     // gate nonlinearity in registers
          int b = m * 16 + off8 + r;
          int len = lenS[b];
          bool valid = t < len;
          float cn = cst[i][r] * sigm(af[r] + 1.0f) + sigm(ai[r]) * tanhf(aj[r]);
          float hn = tanhf(cn) * sigm(ao[r]);
          if (valid) cst[i][r] = cn;
          float hold = (float)hS[b * LDH + col];
          h2buf[i][r] = valid ? hn : hold;          // frozen state past seq_len
          if (col < HDIM) {
            float ov = valid ? hn : 0.0f;
            if (c < 2) {
              psum[i][r] += ov;                     // target pooling
            } else {
              int tout = (fw || !valid) ? t : (len - 1 - t);  // bw re-reversal
              outp[((size_t)b * LC_ + tout) * (2 * HDIM) + coloff + col] = ov;
            }
          }
        }
      }
    }
    __syncthreads();                      // all hS(t-1) reads done
#pragma unroll
    for (int i = 0; i < 3; ++i) {
      int sblk = wv + i * 32;
      if (sblk < 76) {
#pragma unroll
        for (int r = 0; r < 8; ++r)
          hS[(bm[i] * 16 + off8 + r) * LDH + bcol[i]] = (__bf16)h2buf[i][r];
      }
    }
    __syncthreads();                      // hS(t) visible
  }
  if (c < 2) {                            // masked mean for target
#pragma unroll
    for (int i = 0; i < 3; ++i) {
      int sblk = wv + i * 32;
      if (sblk < 76 && bcol[i] < HDIM) {
#pragma unroll
        for (int r = 0; r < 8; ++r) {
          int b = bm[i] * 16 + off8 + r;
          pooled[(size_t)b * (2 * HDIM) + coloff + bcol[i]] = psum[i][r] / (float)lenS[b];
        }
      }
    }
  }
}

// ---------------- K5a: q = pooled @ w_attn^T ; logits = pooled @ w_dense + b --
__global__ void attn_q_kernel(const float* __restrict__ pooled,
                              const float* __restrict__ w_attn,
                              const float* __restrict__ w_dense,
                              const float* __restrict__ b_dense,
                              float* __restrict__ q, float* __restrict__ logits) {
  int tid = blockIdx.x * blockDim.x + threadIdx.x;
  if (tid >= BATCH * 2 * HDIM) return;
  int b = tid / (2 * HDIM), d = tid % (2 * HDIM);
  const float* pb = pooled + (size_t)b * 2 * HDIM;
  float sv = 0.0f;
  for (int e = 0; e < 2 * HDIM; ++e) sv += w_attn[(size_t)d * 2 * HDIM + e] * pb[e];
  q[tid] = sv;
  if (d == 0) {
    float lg = b_dense[0];
    for (int e = 0; e < 2 * HDIM; ++e) lg += pb[e] * w_dense[e];
    logits[b] = lg;
  }
}

// ---------------- K5b: score[b,l] = tanh(h_l . q + b_attn) --------------------
__global__ void score_kernel(const float* __restrict__ h_l, const float* __restrict__ q,
                             const float* __restrict__ b_attn, float* __restrict__ score) {
  int tid = blockIdx.x * blockDim.x + threadIdx.x;
  if (tid >= BATCH * LC_) return;
  int b = tid / LC_;
  const float* hv = h_l + (size_t)tid * 2 * HDIM;
  const float* qb = q + (size_t)b * 2 * HDIM;
  float sv = 0.0f;
  for (int d = 0; d < 2 * HDIM; ++d) sv += hv[d] * qb[d];
  score[tid] = tanhf(sv + b_attn[0]);
}

extern "C" void kernel_launch(void* const* d_in, const int* in_sizes, int n_in,
                              void* d_out, int out_size, void* d_ws, size_t ws_size,
                              hipStream_t stream) {
  (void)in_sizes; (void)n_in; (void)out_size; (void)ws_size;
  const int* left_x     = (const int*)d_in[0];
  const int* left_len   = (const int*)d_in[1];
  const int* target_x   = (const int*)d_in[2];
  const int* target_len = (const int*)d_in[3];
  const int* right_x    = (const int*)d_in[4];
  const int* right_len  = (const int*)d_in[5];
  const float* emb      = (const float*)d_in[6];
  WArgs wargs;
  for (int i = 0; i < 6; ++i) {
    wargs.w[i] = (const float*)d_in[7 + 2 * i];
    wargs.b[i] = (const float*)d_in[8 + 2 * i];
  }
  const float* w_attn  = (const float*)d_in[19];
  const float* b_attn  = (const float*)d_in[20];
  const float* w_dense = (const float*)d_in[21];
  const float* b_dense = (const float*)d_in[22];
  SArgs sargs;
  sargs.ids[0] = target_x; sargs.ids[1] = left_x; sargs.ids[2] = right_x;
  sargs.lens[0] = target_len; sargs.lens[1] = left_len; sargs.lens[2] = right_len;

  char* ws = (char*)d_ws;
  size_t o = 0;
  auto alloc = [&](size_t bytes) {
    void* p = ws + o; o += (bytes + 255) & ~(size_t)255; return p;
  };
  __bf16* WxT    = (__bf16*)alloc((size_t)6 * NG * KP * 2);
  __bf16* WhT    = (__bf16*)alloc((size_t)6 * NG * KP * 2);
  float*  biasB  = (float*) alloc((size_t)6 * NG * 4);
  __bf16* Ab     = (__bf16*)alloc((size_t)NROWS_TOT * KP * 2);
  float*  pre    = (float*) alloc((size_t)NROWS_TOT * NG * 4);
  float*  h_l    = (float*) alloc((size_t)BATCH * LC_ * 2 * HDIM * 4);
  float*  pooled = (float*) alloc((size_t)BATCH * 2 * HDIM * 4);
  float*  q      = (float*) alloc((size_t)BATCH * 2 * HDIM * 4);

  float* logits = (float*)d_out;           // [64]
  float* score  = logits + BATCH;          // [64*96]
  float* h_r    = score + BATCH * LC_;     // [64*96*600]

  int n1 = 6 * NG * KP;
  prep_weights_kernel<<<(n1 + 255) / 256, 256, 0, stream>>>(wargs, WxT, WhT, biasB);
  int n2 = NROWS_TOT * KP;
  prep_embed_kernel<<<(n2 + 255) / 256, 256, 0, stream>>>(sargs, emb, Ab);
  dim3 g3(912, 6);
  xw_gemm_kernel<<<g3, 256, 0, stream>>>(Ab, WxT, biasB, pre);
  lstm_chain_kernel<<<6, 1024, 0, stream>>>(sargs, WhT, pre, pooled, h_l, h_r);
  int n5 = BATCH * 2 * HDIM;
  attn_q_kernel<<<(n5 + 255) / 256, 256, 0, stream>>>(pooled, w_attn, w_dense, b_dense, q, logits);
  int n6 = BATCH * LC_;
  score_kernel<<<(n6 + 255) / 256, 256, 0, stream>>>(h_l, q, b_attn, score);
}
